// DecoderTriplane_39900246179918
// MI455X (gfx1250) — compile-verified, matrix operand order checked by reading, not run on hardware
//
#include <hip/hip_runtime.h>
#include <hip/hip_bf16.h>

typedef __attribute__((ext_vector_type(16))) _Float16 v16h;
typedef __attribute__((ext_vector_type(8)))  float    v8f;

namespace {
constexpr int kNb    = 4;
constexpr int kM     = 262144;
constexpr int kC     = 40;
constexpr int kPlane = 64 * 64;          // H*W
constexpr int kNM    = kNb * kM;         // 1,048,576 points
constexpr int kTiles = kNM / 16;         // 65,536 tiles of 16 points
constexpr int kNW    = 8;                // waves per block
constexpr int kNFrag = 34;               // 16 (L1) + 8 (L2) + 8 (L3) + 2 (L4)
constexpr float kLR  = 4.0f / 7.0f;
}

// A-fragment: per-lane K offsets {0..7} and {16..23} relative to base -> two b128 loads
__device__ __forceinline__ v16h ld_frag_a(const _Float16* p) {
  union { uint4 u[2]; v16h h; } t;
  t.u[0] = *(const uint4*)(p);
  t.u[1] = *(const uint4*)(p + 16);
  return t.h;
}
// B-fragment: 16 contiguous f16 per lane in the pre-swizzled buffer
__device__ __forceinline__ v16h ld_frag_b(const _Float16* p) {
  union { uint4 u[2]; v16h h; } t;
  t.u[0] = *(const uint4*)(p);
  t.u[1] = *(const uint4*)(p + 8);
  return t.h;
}

__device__ __forceinline__ v8f wmma_f16(v16h a, v16h b, v8f c) {
  // D = A(16x32 f16) * B(32x16 f16) + C(16x16 f32)
  return __builtin_amdgcn_wmma_f32_16x16x32_f16(
      /*neg_a=*/false, a, /*neg_b=*/false, b,
      /*c_mod=*/(short)0, c, /*reuse_a=*/false, /*reuse_b=*/false);
}

__device__ __forceinline__ float sigmoidf_(float v) {
  return 1.0f / (1.0f + __expf(-v));
}

__global__ __launch_bounds__(kNW * 32)
void triplane_decoder_kernel(const float* __restrict__ pf,
                             const float* __restrict__ coords,
                             const float* __restrict__ W1, const float* __restrict__ b1,
                             const float* __restrict__ W2, const float* __restrict__ b2,
                             const float* __restrict__ W3, const float* __restrict__ b3,
                             const float* __restrict__ W4, const float* __restrict__ b4,
                             float* __restrict__ out) {
  // fragment store: [frag][lane][16 f16]
  __shared__ __align__(16) _Float16 sW[kNFrag * 512];
  __shared__ __align__(16) _Float16 sA[kNW][16 * 128];   // layer-1 input staging (K padded to 128)
  __shared__ __align__(16) _Float16 sH[kNW][2][16 * 64]; // hidden ping-pong
  __shared__ __align__(16) float    sO[kNW][16 * 4];     // final 16x4 outputs

  // ---- swizzle all weights into WMMA B-fragment layout (once per block) ----
  // frag index map: L1: kc*4+j (0..15), L2: 16+kc*4+j, L3: 24+kc*4+j, L4: 32+kc
  for (int e = threadIdx.x; e < kNFrag * 32 * 8; e += blockDim.x) {
    const int fi = e >> 8;
    const int ln = (e >> 3) & 31;
    const int d  = e & 7;
    int layer, kc, j;
    if (fi < 16)      { layer = 1; kc = fi >> 2;        j = fi & 3; }
    else if (fi < 24) { layer = 2; kc = (fi - 16) >> 2; j = (fi - 16) & 3; }
    else if (fi < 32) { layer = 3; kc = (fi - 24) >> 2; j = (fi - 24) & 3; }
    else              { layer = 4; kc = fi - 32;        j = 0; }
    const int n  = j * 16 + (ln & 15);
    const int k0 = kc * 32 + ((ln & 16) ? 8 : 0) + ((d & 4) ? 16 : 0) + 2 * (d & 3);
    float w0 = 0.f, w1 = 0.f;
    if (layer == 1) {
      if (k0     < 120) w0 = W1[k0 * 64 + n];
      if (k0 + 1 < 120) w1 = W1[(k0 + 1) * 64 + n];
    } else if (layer == 2) {
      w0 = W2[k0 * 64 + n]; w1 = W2[(k0 + 1) * 64 + n];
    } else if (layer == 3) {
      w0 = W3[k0 * 64 + n]; w1 = W3[(k0 + 1) * 64 + n];
    } else {
      if (n < 4) { w0 = W4[k0 * 4 + n]; w1 = W4[(k0 + 1) * 4 + n]; }
    }
    _Float16* dst = sW + fi * 512 + ln * 16 + d * 2;
    dst[0] = (_Float16)w0;
    dst[1] = (_Float16)w1;
  }
  __syncthreads();

  const int wave = threadIdx.x >> 5;
  const int lane = threadIdx.x & 31;
  const int p    = lane & 15;   // point-in-tile (gather) == output column (WMMA C/D)
  const int half = lane >> 4;

  _Float16* stA  = sA[wave];
  _Float16* stHa = sH[wave][0];
  _Float16* stHb = sH[wave][1];
  float*    stO  = sO[wave];

  float b1r[4], b2r[4], b3r[4];
#pragma unroll
  for (int j = 0; j < 4; ++j) {
    b1r[j] = b1[j * 16 + p];
    b2r[j] = b2[j * 16 + p];
    b3r[j] = b3[j * 16 + p];
  }
  const float b4r = (p < 4) ? b4[p] : 0.f;

  for (int tile = blockIdx.x * kNW + wave; tile < kTiles; tile += gridDim.x * kNW) {
    const int P  = tile * 16 + p;
    const int nb = P / kM;
    const float cx = coords[3 * P + 0];
    const float cy = coords[3 * P + 1];
    const float cz = coords[3 * P + 2];

    // ---- per-plane bilinear params (planes: (x,y), (x,z), (z,y)) ----
    float w00[3], w01[3], w10[3], w11[3];
    int   o00[3], o01[3], o10[3], o11[3];
    const float* pb[3];
#pragma unroll
    for (int pl = 0; pl < 3; ++pl) {
      const float gx = (pl == 2) ? cz : cx;
      const float gy = (pl == 1) ? cz : cy;
      const float ix = (gx + 1.f) * 32.f - 0.5f;
      const float iy = (gy + 1.f) * 32.f - 0.5f;
      const float x0f = floorf(ix), y0f = floorf(iy);
      const float wx = ix - x0f, wy = iy - y0f;
      const int x0 = (int)x0f, y0 = (int)y0f;
      const bool vx0 = (x0 >=  0) & (x0 < 64);
      const bool vx1 = (x0 >= -1) & (x0 < 63);
      const bool vy0 = (y0 >=  0) & (y0 < 64);
      const bool vy1 = (y0 >= -1) & (y0 < 63);
      w00[pl] = (vx0 & vy0) ? (1.f - wx) * (1.f - wy) : 0.f;
      w01[pl] = (vx1 & vy0) ? wx * (1.f - wy)         : 0.f;
      w10[pl] = (vx0 & vy1) ? (1.f - wx) * wy         : 0.f;
      w11[pl] = (vx1 & vy1) ? wx * wy                 : 0.f;
      const int x0c = min(max(x0, 0), 63), x1c = min(max(x0 + 1, 0), 63);
      const int y0c = min(max(y0, 0), 63), y1c = min(max(y0 + 1, 0), 63);
      o00[pl] = y0c * 64 + x0c; o01[pl] = y0c * 64 + x1c;
      o10[pl] = y1c * 64 + x0c; o11[pl] = y1c * 64 + x1c;
      pb[pl]  = pf + (size_t)((nb * 3 + pl) * kC) * kPlane;
    }

    // ---- gather 120 features -> LDS staging (lane halves split channels) ----
    _Float16* rowA = stA + p * 128;
#pragma unroll
    for (int pl = 0; pl < 3; ++pl) {
      int c0, cnt;
      if (pl == 0)      { c0 = 0;              cnt = half ? 0 : 40; }
      else if (pl == 1) { c0 = half ? 20 : 0;  cnt = 20; }
      else              { c0 = 0;              cnt = half ? 40 : 0; }
      const float* bp = pb[pl];
      for (int c = c0; c < c0 + cnt; ++c) {
        const float* cp = bp + c * kPlane;
        const float v = w00[pl] * cp[o00[pl]] + w01[pl] * cp[o01[pl]]
                      + w10[pl] * cp[o10[pl]] + w11[pl] * cp[o11[pl]];
        rowA[pl * 40 + c] = (_Float16)v;
      }
    }
    if (!half) { *(uint4*)(rowA + 120) = make_uint4(0u, 0u, 0u, 0u); }
    asm volatile("s_wait_dscnt 0" ::: "memory");

    // ---- layer 1: (16x128) @ (128x64), K-chunks of 32, 4 N-tiles ----
    {
      const _Float16* arow = stA + p * 128 + (half ? 8 : 0);
      v16h a[4];
#pragma unroll
      for (int kc = 0; kc < 4; ++kc) a[kc] = ld_frag_a(arow + kc * 32);
#pragma unroll
      for (int j = 0; j < 4; ++j) {
        v8f c;
#pragma unroll
        for (int r = 0; r < 8; ++r) c[r] = b1r[j];
#pragma unroll
        for (int kc = 0; kc < 4; ++kc)
          c = wmma_f16(a[kc], ld_frag_b(sW + (kc * 4 + j) * 512 + lane * 16), c);
#pragma unroll
        for (int r = 0; r < 8; ++r) {
          const float v = fmaxf(c[r], 0.f);
          stHa[(half ? r + 8 : r) * 64 + j * 16 + p] = (_Float16)v;
        }
      }
      asm volatile("s_wait_dscnt 0" ::: "memory");
    }

    // ---- layer 2: stHa -> stHb ----
    {
      const _Float16* hrow = stHa + p * 64 + (half ? 8 : 0);
      const v16h h0 = ld_frag_a(hrow);
      const v16h h1 = ld_frag_a(hrow + 32);
#pragma unroll
      for (int j = 0; j < 4; ++j) {
        v8f c;
#pragma unroll
        for (int r = 0; r < 8; ++r) c[r] = b2r[j];
        c = wmma_f16(h0, ld_frag_b(sW + (16 + j) * 512 + lane * 16), c);
        c = wmma_f16(h1, ld_frag_b(sW + (20 + j) * 512 + lane * 16), c);
#pragma unroll
        for (int r = 0; r < 8; ++r) {
          const float v = fmaxf(c[r], 0.f);
          stHb[(half ? r + 8 : r) * 64 + j * 16 + p] = (_Float16)v;
        }
      }
      asm volatile("s_wait_dscnt 0" ::: "memory");
    }

    // ---- layer 3: stHb -> stHa ----
    {
      const _Float16* hrow = stHb + p * 64 + (half ? 8 : 0);
      const v16h h0 = ld_frag_a(hrow);
      const v16h h1 = ld_frag_a(hrow + 32);
#pragma unroll
      for (int j = 0; j < 4; ++j) {
        v8f c;
#pragma unroll
        for (int r = 0; r < 8; ++r) c[r] = b3r[j];
        c = wmma_f16(h0, ld_frag_b(sW + (24 + j) * 512 + lane * 16), c);
        c = wmma_f16(h1, ld_frag_b(sW + (28 + j) * 512 + lane * 16), c);
#pragma unroll
        for (int r = 0; r < 8; ++r) {
          const float v = fmaxf(c[r], 0.f);
          stHa[(half ? r + 8 : r) * 64 + j * 16 + p] = (_Float16)v;
        }
      }
      asm volatile("s_wait_dscnt 0" ::: "memory");
    }

    // ---- layer 4: (16x64) @ (64x4 padded to 16) ----
    {
      const _Float16* hrow = stHa + p * 64 + (half ? 8 : 0);
      const v16h h0 = ld_frag_a(hrow);
      const v16h h1 = ld_frag_a(hrow + 32);
      v8f c;
#pragma unroll
      for (int r = 0; r < 8; ++r) c[r] = b4r;
      c = wmma_f16(h0, ld_frag_b(sW + 32 * 512 + lane * 16), c);
      c = wmma_f16(h1, ld_frag_b(sW + 33 * 512 + lane * 16), c);
      if (p < 4) {
#pragma unroll
        for (int r = 0; r < 8; ++r)
          stO[(half ? r + 8 : r) * 4 + p] = c[r];
      }
      asm volatile("s_wait_dscnt 0" ::: "memory");
    }

    // ---- epilogue: lanes 0..15 each own one point (coords still live) ----
    if (lane < 16) {
      const float4 o = *(const float4*)(stO + lane * 4);
      const bool outside = (fabsf(cx) > kLR) | (fabsf(cy) > kLR) | (fabsf(cz) > kLR);
      const float sig = outside ? 0.f : fmaxf(o.x, 0.f);
      const float r = sigmoidf_(o.y) * 1.002f - 0.001f;
      const float g = sigmoidf_(o.z) * 1.002f - 0.001f;
      const float b = sigmoidf_(o.w) * 1.002f - 0.001f;
      out[3 * P + 0] = r;
      out[3 * P + 1] = g;
      out[3 * P + 2] = b;
      out[3 * kNM + P] = sig;
    }
  }
}

extern "C" void kernel_launch(void* const* d_in, const int* in_sizes, int n_in,
                              void* d_out, int out_size, void* d_ws, size_t ws_size,
                              hipStream_t stream) {
  (void)in_sizes; (void)n_in; (void)out_size; (void)d_ws; (void)ws_size;
  const float* pf     = (const float*)d_in[0];
  const float* coords = (const float*)d_in[1];
  const float* W1 = (const float*)d_in[2];
  const float* b1 = (const float*)d_in[3];
  const float* W2 = (const float*)d_in[4];
  const float* b2 = (const float*)d_in[5];
  const float* W3 = (const float*)d_in[6];
  const float* b3 = (const float*)d_in[7];
  const float* W4 = (const float*)d_in[8];
  const float* b4 = (const float*)d_in[9];
  float* out = (float*)d_out;

  dim3 grid(2048);           // 16384 waves, 4 tiles each over 65536 tiles
  dim3 block(kNW * 32);      // 8 waves of 32
  triplane_decoder_kernel<<<grid, block, 0, stream>>>(
      pf, coords, W1, b1, W2, b2, W3, b3, W4, b4, out);
}
